// Llama4TextMoe_55035710931369
// MI455X (gfx1250) — matrix operand order repaired and checked by reading, not compiled, not run on hardware
//
#include <hip/hip_runtime.h>
#include <hip/hip_bf16.h>

// ---------------------------------------------------------------------------
// MI455X (gfx1250, wave32) Llama4 MoE: top-1 routed + shared expert, f16 WMMA
// + Tensor Data Mover for f16 activation tiles (TENSORcnt path)
// ---------------------------------------------------------------------------

typedef __attribute__((ext_vector_type(16))) _Float16 v16h;
typedef __attribute__((ext_vector_type(8)))  _Float16 v8h;
typedef __attribute__((ext_vector_type(2)))  __fp16   v2fp16;
typedef __attribute__((ext_vector_type(8)))  float    v8f;
typedef unsigned int u32x4 __attribute__((ext_vector_type(4)));
typedef int          i32x8 __attribute__((ext_vector_type(8)));
typedef int          i32x4 __attribute__((ext_vector_type(4)));

#define BM 128
#define BN 128
#define BK 32
#define KSTRIDE 40   // padded LDS k-stride (elems): 40*2B = 80B => conflict-free row map

__device__ __forceinline__ unsigned short f2h_bits(float f) {
  union { _Float16 h; unsigned short u; } c;
  c.h = (_Float16)f;
  return c.u;
}

// packed f32x2 -> f16x2 (v_cvt_pk_rtz_f16_f32)
__device__ __forceinline__ unsigned pk2h(float x, float y) {
  union { v2fp16 v; unsigned u; } c;
  c.v = __builtin_amdgcn_cvt_pkrtz(x, y);
  return c.u;
}

// Tensor Data Mover: 2D f16 tile (tile_w x tile_h), row stride `stride` elems,
// into LDS at lds_byte_off with 16B pad after each 64B row (=> KSTRIDE 40).
__device__ __forceinline__ void tdm_load_2d_f16(
    unsigned lds_byte_off, const void* gsrc,
    int tensor_w, int tensor_h, int tile_w, int tile_h, int stride) {
  unsigned long long ga = (unsigned long long)(size_t)gsrc;
  u32x4 g0;
  g0[0] = 1u;                                    // count=1 (valid user descriptor)
  g0[1] = lds_byte_off;                          // lds_addr
  g0[2] = (unsigned)(ga & 0xFFFFFFFFu);          // global_addr[31:0]
  g0[3] = (unsigned)((ga >> 32) & 0x1FFFFFFu)    // global_addr[56:32]
        | (2u << 30);                            // type = 2 ("image")
  i32x8 g1;
  g1[0] = (int)((1u << 16)                       // data_size = 2 bytes
              | (1u << 20)                       // pad_enable
              | (2u << 22)                       // pad_interval: 8 DWORDs (64B)
              | (3u << 25));                     // pad_amount: 4 DWORDs (16B)
  g1[1] = (int)(((unsigned)tensor_w & 0xFFFFu) << 16);                     // tensor_dim0 lo
  g1[2] = (int)(((unsigned)tensor_w >> 16) |
                (((unsigned)tensor_h & 0xFFFFu) << 16));                   // dim0 hi | dim1 lo
  g1[3] = (int)(((unsigned)tensor_h >> 16) |
                (((unsigned)tile_w & 0xFFFFu) << 16));                     // dim1 hi | tile_dim0
  g1[4] = (int)((unsigned)tile_h & 0xFFFFu);                               // tile_dim1
  g1[5] = (int)(unsigned)stride;                                          // tensor_dim0_stride lo
  g1[6] = 0;
  g1[7] = 0;
  i32x4 z4; z4[0] = 0; z4[1] = 0; z4[2] = 0; z4[3] = 0;                    // 2D: groups 2/3 unused
  i32x8 z8;
  #pragma unroll
  for (int i = 0; i < 8; ++i) z8[i] = 0;
  __builtin_amdgcn_tensor_load_to_lds(g0, g1, z4, z4, z8, 0);              // 6-arg (clang-23)
}

// ---------------------------- small helper kernels -------------------------

__global__ void init_small(int* cnt, int* cursor, int E) {
  int i = threadIdx.x;
  if (i < E) { cnt[i] = 0; cursor[i] = 0; }
}

// one wave32 per token: 16 dot products of length H, argmax, sigmoid
__global__ __launch_bounds__(256) void router_kernel(
    const float* __restrict__ x, const float* __restrict__ w,
    int* __restrict__ eid, float* __restrict__ esc, int T, int H) {
  int wid  = threadIdx.x >> 5;
  int lane = threadIdx.x & 31;
  int t = blockIdx.x * 8 + wid;
  if (t >= T) return;
  const float* xr = x + (size_t)t * H;
  float acc[16];
  #pragma unroll
  for (int e = 0; e < 16; ++e) acc[e] = 0.f;
  for (int h = lane; h < H; h += 32) {
    float xv = xr[h];
    #pragma unroll
    for (int e = 0; e < 16; ++e) acc[e] += xv * w[e * H + h];
  }
  #pragma unroll
  for (int e = 0; e < 16; ++e) {
    #pragma unroll
    for (int off = 16; off > 0; off >>= 1)
      acc[e] += __shfl_xor(acc[e], off, 32);
  }
  if (lane == 0) {
    int best = 0; float bv = acc[0];
    #pragma unroll
    for (int e = 1; e < 16; ++e) if (acc[e] > bv) { bv = acc[e]; best = e; }
    eid[t] = best;
    esc[t] = 1.f / (1.f + __expf(-bv));
  }
}

__global__ void hist_kernel(const int* __restrict__ eid, int* cnt, int T) {
  int t = blockIdx.x * 256 + threadIdx.x;
  if (t < T) atomicAdd(&cnt[eid[t]], 1);
}

// single-thread scan + tile table: 128-row GEMM tiles never straddle experts
__global__ void scan_kernel(const int* __restrict__ cnt, int* off, int* ntiles,
                            int* te, int* tr0, int* trn, int E) {
  if (threadIdx.x != 0 || blockIdx.x != 0) return;
  int run = 0, nt = 0;
  for (int e = 0; e < E; ++e) {
    off[e] = run;
    int c = cnt[e];
    for (int p = 0; p < c; p += BM) {
      te[nt] = e; tr0[nt] = run + p;
      int r = c - p; trn[nt] = (r < BM) ? r : BM;
      ++nt;
    }
    run += c;
  }
  off[E] = run;
  *ntiles = nt;
}

__global__ void scatter_kernel(const int* __restrict__ eid, const int* __restrict__ off,
                               int* cursor, int* __restrict__ perm, int T) {
  int t = blockIdx.x * 256 + threadIdx.x;
  if (t < T) {
    int e = eid[t];
    int p = atomicAdd(&cursor[e], 1);
    perm[off[e] + p] = t;
  }
}

// h[t,i] = silu(gate) * up, emitted as f16 to feed the down-proj GEMM directly
__global__ __launch_bounds__(256) void act_kernel(
    const float* __restrict__ C, unsigned short* __restrict__ h, int T, int I) {
  size_t idx = (size_t)blockIdx.x * 256 + threadIdx.x;
  size_t total = (size_t)T * I;
  if (idx >= total) return;
  size_t t = idx / I;
  size_t i = idx - t * (size_t)I;
  const float* row = C + t * (size_t)(2 * I);
  float g = row[i];
  float u = row[I + i];
  float s = g / (1.f + __expf(-g));
  h[idx] = f2h_bits(s * u);
}

// ------------------------------- WMMA GEMM ---------------------------------
// C[M,N] (+)= A[M,K] @ B[K,N], f16 multiply / f32 accumulate.
//  A_F16: A is f16 in ws -> tiles staged by the Tensor Data Mover (TENSORcnt)
//  A_GATHER: A rows gathered via perm[] and scaled by score[] (routed gate_up)
//  B_COL: B stored column-major in K (weight rows contiguous in K)
//  C_ADD / C_SCATTER: epilogue accumulate / row scatter via perm[]
//  USE_TABLE: M-tiles come from device-built expert tile table
template<int A_F16, int A_GATHER, int B_COL, int C_ADD, int C_SCATTER, int USE_TABLE>
__global__ __launch_bounds__(256) void gemm_wmma_kernel(
    const void* __restrict__ Av, int lda,
    const float* __restrict__ B0, long ldb, long bexp_stride,
    float* __restrict__ Cp, int ldc,
    int M, int N, int K,
    const int* __restrict__ tile_e, const int* __restrict__ tile_r0,
    const int* __restrict__ tile_rows, const int* __restrict__ ntiles,
    const int* __restrict__ perm, const float* __restrict__ score) {
  __shared__ unsigned short Ash[2][BM * KSTRIDE];
  __shared__ unsigned short Bsh[2][BN * KSTRIDE];

  const int tid = threadIdx.x;
  int row0, rows, expert;
  if (USE_TABLE) {
    int ti = blockIdx.y;
    if (ti >= *ntiles) return;        // uniform per block: safe early exit
    expert = tile_e[ti]; row0 = tile_r0[ti]; rows = tile_rows[ti];
    if (rows <= 0) return;
  } else {
    expert = 0; row0 = blockIdx.y * BM; rows = BM;
  }
  const int bn = blockIdx.x * BN;
  const float* __restrict__ Bp = B0 + (long)expert * bexp_stride;

  // wave32 coordinates: 8 waves in a 2(M) x 4(N) grid, 64x32 per wave
  const int lane  = tid & 31;
  const int wid   = tid >> 5;
  const int mo    = (wid >> 2) * 64;
  const int no    = (wid & 3) * 32;
  const int lrow  = lane & 15;
  const int lhalf = lane >> 4;
  const int abase = lhalf * 16;   // A frag: K chunk  [abase..+15]
  const int bkoff = lhalf * 16;   // B frag: K chunk  [bkoff..+15]

  v8f acc[4][2];
  #pragma unroll
  for (int i = 0; i < 4; ++i)
    #pragma unroll
    for (int j = 0; j < 2; ++j)
      #pragma unroll
      for (int r = 0; r < 8; ++r) acc[i][j][r] = 0.f;

  float4 aF[4];
  float4 bF[4];

  auto tdmA = [&](int kt, int buf) {   // whole A tile via Tensor Data Mover
    tdm_load_2d_f16((unsigned)(size_t)&Ash[buf][0],
                    (const unsigned short*)Av + (size_t)row0 * lda + kt,
                    lda, rows, BK, rows, lda);
  };

  auto fetchA = [&](int kt) {          // fp32 A, optional gather+score scaling
    const float* A = (const float*)Av;
    #pragma unroll
    for (int i = 0; i < 4; ++i) {
      int idx = tid + 256 * i;
      int r = idx >> 3, c4 = idx & 7;
      bool valid = (!USE_TABLE) || (r < rows);
      float sc = 1.f;
      size_t grow;
      if (A_GATHER) {
        int pr = valid ? perm[row0 + r] : 0;
        grow = (size_t)pr;
        sc = valid ? score[pr] : 0.f;
      } else {
        grow = (size_t)(row0 + r);
      }
      float4 v = {0.f, 0.f, 0.f, 0.f};
      if (valid) v = *(const float4*)&A[grow * (size_t)lda + kt + c4 * 4];
      v.x *= sc; v.y *= sc; v.z *= sc; v.w *= sc;
      aF[i] = v;
    }
  };

  auto fetchB = [&](int kt) {
    if (B_COL) {
      if (kt + 2 * BK < K) __builtin_prefetch(&Bp[(size_t)bn * ldb + kt + 2 * BK], 0, 1);
      #pragma unroll
      for (int i = 0; i < 4; ++i) {
        int idx = tid + 256 * i;
        int n = idx >> 3, c4 = idx & 7;
        bF[i] = *(const float4*)&Bp[(size_t)(bn + n) * ldb + kt + c4 * 4];
      }
    } else {
      if (kt + 2 * BK < K) __builtin_prefetch(&Bp[(size_t)(kt + 2 * BK) * ldb + bn], 0, 1);
      #pragma unroll
      for (int i = 0; i < 4; ++i) {
        int idx = tid + 256 * i;
        int k = idx >> 5, c4 = idx & 31;
        bF[i] = *(const float4*)&Bp[(size_t)(kt + k) * ldb + bn + c4 * 4];
      }
    }
  };

  auto stageA = [&](int buf) {         // fp32 -> f16 pack -> LDS
    #pragma unroll
    for (int i = 0; i < 4; ++i) {
      int idx = tid + 256 * i;
      int r = idx >> 3, c4 = idx & 7;
      uint2 pk;
      pk.x = pk2h(aF[i].x, aF[i].y);
      pk.y = pk2h(aF[i].z, aF[i].w);
      *(uint2*)&Ash[buf][r * KSTRIDE + c4 * 4] = pk;
    }
  };

  auto stageB = [&](int buf) {
    if (B_COL) {
      #pragma unroll
      for (int i = 0; i < 4; ++i) {
        int idx = tid + 256 * i;
        int n = idx >> 3, c4 = idx & 7;
        uint2 pk;
        pk.x = pk2h(bF[i].x, bF[i].y);
        pk.y = pk2h(bF[i].z, bF[i].w);
        *(uint2*)&Bsh[buf][n * KSTRIDE + c4 * 4] = pk;
      }
    } else {
      // transpose (k,n) -> LDS [n][k]
      #pragma unroll
      for (int i = 0; i < 4; ++i) {
        int idx = tid + 256 * i;
        int k = idx >> 5, c4 = idx & 31;
        Bsh[buf][(c4 * 4 + 0) * KSTRIDE + k] = f2h_bits(bF[i].x);
        Bsh[buf][(c4 * 4 + 1) * KSTRIDE + k] = f2h_bits(bF[i].y);
        Bsh[buf][(c4 * 4 + 2) * KSTRIDE + k] = f2h_bits(bF[i].z);
        Bsh[buf][(c4 * 4 + 3) * KSTRIDE + k] = f2h_bits(bF[i].w);
      }
    }
  };

  auto compute = [&](int buf) {
    v16h afr[4];
    #pragma unroll
    for (int i = 0; i < 4; ++i) {
      const unsigned short* ap = &Ash[buf][(mo + i * 16 + lrow) * KSTRIDE];
      v8h lo = *(const v8h*)&ap[abase];
      v8h hi = *(const v8h*)&ap[abase + 8];
      union { v16h v; v8h h[2]; } u;
      u.h[0] = lo; u.h[1] = hi;
      afr[i] = u.v;
    }
    v16h bfr[2];
    #pragma unroll
    for (int j = 0; j < 2; ++j)
      bfr[j] = *(const v16h*)&Bsh[buf][(no + j * 16 + lrow) * KSTRIDE + bkoff];
    #pragma unroll
    for (int i = 0; i < 4; ++i)
      #pragma unroll
      for (int j = 0; j < 2; ++j)
        acc[i][j] = __builtin_amdgcn_wmma_f32_16x16x32_f16(
            false, afr[i], false, bfr[j], (short)0, acc[i][j], false, false);
  };

  // prologue
  if constexpr (A_F16) {
    if (tid == 0) tdmA(0, 0);          // one DMA descriptor per tile (EXEC-independent)
  } else {
    fetchA(0);
  }
  fetchB(0);
  if constexpr (!A_F16) stageA(0);
  stageB(0);
  if constexpr (A_F16) {
    if (tid < 32) __builtin_amdgcn_s_wait_tensorcnt(0);
  }
  __syncthreads();

  int cur = 0;
  for (int kt = 0; kt < K; kt += BK) {
    const bool more = (kt + BK) < K;
    if (more) {
      if constexpr (A_F16) {
        if (tid == 0) tdmA(kt + BK, cur ^ 1);   // nxt buf free: barrier'd last iter
      } else {
        fetchA(kt + BK);
      }
      fetchB(kt + BK);
    }
    compute(cur);
    __syncthreads();
    if (more) {
      if constexpr (!A_F16) stageA(cur ^ 1);
      stageB(cur ^ 1);
      if constexpr (A_F16) {
        if (tid < 32) __builtin_amdgcn_s_wait_tensorcnt(0);
      }
      __syncthreads();
      cur ^= 1;
    }
  }

  // epilogue: C/D layout -> lane {0-15,16-31} = M {r, r+8}, N = lane&15
  #pragma unroll
  for (int i = 0; i < 4; ++i) {
    #pragma unroll
    for (int j = 0; j < 2; ++j) {
      #pragma unroll
      for (int r = 0; r < 8; ++r) {
        int mrow = mo + i * 16 + lhalf * 8 + r;
        if (USE_TABLE && mrow >= rows) continue;
        int ncol = bn + no + j * 16 + lrow;
        size_t orow = C_SCATTER ? (size_t)perm[row0 + mrow] : (size_t)(row0 + mrow);
        float* dst = &Cp[orow * (size_t)ldc + ncol];
        float v = acc[i][j][r];
        if (C_ADD) *dst += v; else *dst = v;
      }
    }
  }
}

// --------------------------------- launcher --------------------------------

extern "C" void kernel_launch(void* const* d_in, const int* in_sizes, int n_in,
                              void* d_out, int out_size, void* d_ws, size_t ws_size,
                              hipStream_t stream) {
  const float* x   = (const float*)d_in[0];  // (T, H)
  const float* rw  = (const float*)d_in[1];  // (E, H)
  const float* gup = (const float*)d_in[2];  // (E, H, 2I)
  const float* dwn = (const float*)d_in[3];  // (E, I, H)
  const float* sgu = (const float*)d_in[4];  // (2I, H)
  const float* sdn = (const float*)d_in[5];  // (H, I)
  float* out = (float*)d_out;                // (T, H)

  const int E = 16;
  const int H = in_sizes[1] / E;
  const int T = in_sizes[0] / H;
  const int I = in_sizes[5] / H;
  const int twoI = 2 * I;
  (void)n_in; (void)out_size; (void)ws_size;

  char* w = (char*)d_ws;
  auto alloc = [&](size_t bytes) {
    char* p = w;
    w += (bytes + 255) & ~(size_t)255;
    return p;
  };
  int*   eid    = (int*)  alloc((size_t)T * 4);
  float* score  = (float*)alloc((size_t)T * 4);
  int*   cnt    = (int*)  alloc(E * 4);
  int*   off    = (int*)  alloc((E + 1) * 4);
  int*   cursor = (int*)  alloc(E * 4);
  int*   ntil   = (int*)  alloc(4);
  int*   te     = (int*)  alloc(64 * 4);
  int*   tr0    = (int*)  alloc(64 * 4);
  int*   trn    = (int*)  alloc(64 * 4);
  int*   perm   = (int*)  alloc((size_t)T * 4);
  float* Cbuf   = (float*)alloc((size_t)T * twoI * 4);       // gate_up output (reused)
  unsigned short* hbuf = (unsigned short*)alloc((size_t)T * I * 2);  // f16 activations

  // routing
  init_small<<<1, 64, 0, stream>>>(cnt, cursor, E);
  router_kernel<<<T / 8, 256, 0, stream>>>(x, rw, eid, score, T, H);
  hist_kernel<<<(T + 255) / 256, 256, 0, stream>>>(eid, cnt, T);
  scan_kernel<<<1, 32, 0, stream>>>(cnt, off, ntil, te, tr0, trn, E);
  scatter_kernel<<<(T + 255) / 256, 256, 0, stream>>>(eid, off, cursor, perm, T);

  dim3 blk(256);
  const int actGrid = (int)(((size_t)T * I + 255) / 256);
  const int maxTiles = T / BM + E;

  // shared expert: C = x @ sgu^T  (B col-major in K)
  gemm_wmma_kernel<0, 0, 1, 0, 0, 0><<<dim3(twoI / BN, T / BM), blk, 0, stream>>>(
      x, H, sgu, (long)H, 0L, Cbuf, twoI, T, twoI, H,
      nullptr, nullptr, nullptr, nullptr, nullptr, nullptr);
  act_kernel<<<actGrid, 256, 0, stream>>>(Cbuf, hbuf, T, I);
  // out = h @ sdn^T   (first writer: plain store; A tiles via TDM)
  gemm_wmma_kernel<1, 0, 1, 0, 0, 0><<<dim3(H / BN, T / BM), blk, 0, stream>>>(
      hbuf, I, sdn, (long)I, 0L, out, H, T, H, I,
      nullptr, nullptr, nullptr, nullptr, nullptr, nullptr);

  // routed experts (top-1, expert-grouped tiles): C = (gather(x)*score) @ gup[e]
  gemm_wmma_kernel<0, 1, 0, 0, 0, 1><<<dim3(twoI / BN, maxTiles), blk, 0, stream>>>(
      x, H, gup, (long)twoI, (long)H * twoI, Cbuf, twoI, T, twoI, H,
      te, tr0, trn, ntil, perm, score);
  act_kernel<<<actGrid, 256, 0, stream>>>(Cbuf, hbuf, T, I);
  // out[perm[row]] += h @ dwn[e]   (A tiles via TDM)
  gemm_wmma_kernel<1, 0, 0, 1, 1, 1><<<dim3(H / BN, maxTiles), blk, 0, stream>>>(
      hbuf, I, dwn, (long)H, (long)I * H, out, H, T, H, I,
      te, tr0, trn, ntil, perm, score);
}
